// TransformerGraphAttention_90125593739479
// MI455X (gfx1250) — matrix operand order repaired and checked by reading, not compiled, not run on hardware
//
#include <hip/hip_runtime.h>
#include <hip/hip_bf16.h>

// TransformerGraphAttention forward for MI455X (gfx1250, wave32, WMMA).
//   B=8, N=2048, D=128.  HBM floor ~= adj (134MB) => flash-attention single
//   pass over adj; all GEMMs on v_wmma_f32_16x16x32_f16 (f16 in, f32 acc).

#define BATCH 8
#define SEQ   2048
#define DIM   128
#define NEG_BIG (-10000.0f)

typedef __attribute__((ext_vector_type(16))) _Float16 v16h;
typedef __attribute__((ext_vector_type(8)))  _Float16 v8h;
typedef __attribute__((ext_vector_type(8)))  float    v8f;

// A-matrix (16x32 f16) fragment from a row-major f16 row pointer.
// lane: row = lane&15; half h -> col = (h&7) + 8*(h>>3) + 16*(lane>>4)
static __device__ inline v16h load_a_frag(const _Float16* rowp, int hi) {
  v8h g0 = *(const v8h*)(rowp + 16 * hi);
  v8h g1 = *(const v8h*)(rowp + 16 * hi + 8);
  v16h a;
#pragma unroll
  for (int i = 0; i < 8; ++i) { a[i] = g0[i]; a[8 + i] = g1[i]; }
  return a;
}

static __device__ inline v8f wmma_f16(v16h a, v16h b, v8f c) {
  return __builtin_amdgcn_wmma_f32_16x16x32_f16(false, a, false, b,
                                                (short)0, c, false, false);
}

// ---------------------------------------------------------------------------
// Kernel 1: QKV projection. q,k stored row-major f16; v stored transposed
// (vT[b][col][row]) so attention's PV B-fragments are contiguous per lane.
// M is a compile-time matrix selector so store addressing / bias pointers are
// branch-free.
// ---------------------------------------------------------------------------
template <int M>
static __device__ inline void project_store(
    const v16h afrag[4], const _Float16* __restrict__ wt,
    const float* __restrict__ biasv,
    _Float16* __restrict__ qh, _Float16* __restrict__ kh,
    _Float16* __restrict__ vT,
    int b, int row0, int lo, int hi)
{
#pragma unroll
  for (int ct = 0; ct < 8; ++ct) {                    // output col tiles
    // Preload all 4 B-fragments so ds_loads overlap the WMMA chain.
    v16h bf[4];
#pragma unroll
    for (int c = 0; c < 4; ++c)
      bf[c] = *(const v16h*)&wt[(16 * ct + lo) * DIM + 32 * c + 16 * hi];
    v8f acc = {};
#pragma unroll
    for (int c = 0; c < 4; ++c) acc = wmma_f16(afrag[c], bf[c], acc);

    const float bval = biasv[16 * ct + lo];
#pragma unroll
    for (int r = 0; r < 8; ++r) {
      const int row = row0 + r + 8 * hi;              // C layout: M = r + 8*hi
      const int col = 16 * ct + lo;                   //           N = lo
      const _Float16 hv = (_Float16)(acc[r] + bval);
      if (M == 0)      qh[((size_t)b * SEQ + row) * DIM + col] = hv;
      else if (M == 1) kh[((size_t)b * SEQ + row) * DIM + col] = hv;
      else             vT[((size_t)b * DIM + col) * SEQ + row] = hv;
    }
  }
}

static __device__ inline void stage_wt(const float* __restrict__ W,
                                       _Float16* __restrict__ wt, int tid) {
  __syncthreads();
  for (int idx = tid; idx < DIM * DIM; idx += 256) {
    const int d = idx >> 7, col = idx & 127;          // idx = d*128 + col
    wt[col * DIM + d] = (_Float16)W[idx];             // store transposed
  }
  __syncthreads();
}

__global__ __launch_bounds__(256) void qkv_kernel(
    const float* __restrict__ inputs,
    const float* __restrict__ Wq, const float* __restrict__ bq,
    const float* __restrict__ Wk, const float* __restrict__ bk,
    const float* __restrict__ Wv, const float* __restrict__ bv,
    _Float16* __restrict__ qh, _Float16* __restrict__ kh,
    _Float16* __restrict__ vT)
{
  __shared__ _Float16 wt[DIM * DIM];  // W transposed: wt[col*128 + d]

  const int tid  = threadIdx.x;
  const int wave = tid >> 5;
  const int lane = tid & 31;
  const int lo   = lane & 15;
  const int hi   = lane >> 4;

  const int gtile = blockIdx.x * 8 + wave;   // global 16-row tile id
  const int b     = gtile >> 7;              // 128 tiles per batch
  const int row0  = (gtile & 127) * 16;

  // Inputs tile 16x128 -> 4 A-fragments (K chunks of 32), fp32 -> f16.
  v16h afrag[4];
  {
    const float* ip = inputs + ((size_t)b * SEQ + row0 + lo) * DIM;
#pragma unroll
    for (int c = 0; c < 4; ++c) {
      v16h a;
#pragma unroll
      for (int i = 0; i < 8; ++i) {
        a[i]     = (_Float16)ip[32 * c + 16 * hi + i];
        a[8 + i] = (_Float16)ip[32 * c + 16 * hi + 8 + i];
      }
      afrag[c] = a;
    }
  }

  stage_wt(Wq, wt, tid);
  project_store<0>(afrag, wt, bq, qh, kh, vT, b, row0, lo, hi);
  stage_wt(Wk, wt, tid);
  project_store<1>(afrag, wt, bk, qh, kh, vT, b, row0, lo, hi);
  stage_wt(Wv, wt, tid);
  project_store<2>(afrag, wt, bv, qh, kh, vT, b, row0, lo, hi);
}

// ---------------------------------------------------------------------------
// Kernel 2: flash attention + adjacency bias + ELU. One wave per 16 output
// rows, online softmax over 64 column-tiles of 32. LDS only for the
// C-layout -> A-layout transpose of P (1KB per wave). adj (the sole HBM
// stream) is prefetched one iteration ahead with global_prefetch_b8.
// ---------------------------------------------------------------------------
__global__ __launch_bounds__(256) void attn_kernel(
    const _Float16* __restrict__ qh, const _Float16* __restrict__ kh,
    const _Float16* __restrict__ vT, const int* __restrict__ adj,
    float* __restrict__ out)
{
  __shared__ _Float16 plds[8 * 16 * 32];  // per-wave 16x32 f16 P tile

  const int tid  = threadIdx.x;
  const int wave = tid >> 5;
  const int lane = tid & 31;
  const int lo   = lane & 15;
  const int hi   = lane >> 4;

  const int b    = blockIdx.x >> 4;                  // 16 blocks per batch
  const int row0 = (blockIdx.x & 15) * 128 + wave * 16;

  // Q fragments for this 16-row tile (K = 128 -> 4 chunks of 32).
  v16h qfrag[4];
  {
    const _Float16* qp = qh + ((size_t)b * SEQ + row0 + lo) * DIM;
#pragma unroll
    for (int c = 0; c < 4; ++c) qfrag[c] = load_a_frag(qp + 32 * c, hi);
  }

  v8f o[8];
#pragma unroll
  for (int ct = 0; ct < 8; ++ct) o[ct] = (v8f){};
  float mrow[8], lrow[8];
#pragma unroll
  for (int r = 0; r < 8; ++r) { mrow[r] = -3.0e38f; lrow[r] = 0.0f; }

  const float scale = 0.08838834764831845f;          // 1/sqrt(128)
  const int*  adjb  = adj + (size_t)b * SEQ * SEQ;
  _Float16*   myp   = plds + wave * (16 * 32);

  for (int j = 0; j < SEQ; j += 32) {
    // ---- prefetch next iteration's adj rows (streaming HBM traffic) ----
    if (j + 32 < SEQ) {
#pragma unroll
      for (int r = 0; r < 8; ++r) {
        const int* anext = adjb + (size_t)(row0 + r + 8 * hi) * SEQ + j + 32;
        __builtin_prefetch(anext + lo, 0, 1);        // global_prefetch_b8
        __builtin_prefetch(anext + 16 + lo, 0, 1);
      }
    }

    // ---- S = Q @ K^T for 32 columns (two 16x16 subtiles) ----
    // Preload the 8 K B-fragments so all global loads are in flight.
    v16h kf[8];
    {
      const _Float16* kbase = kh + ((size_t)b * SEQ + j) * DIM;
#pragma unroll
      for (int c = 0; c < 4; ++c) {
        kf[c]     = *(const v16h*)(kbase + (size_t)lo * DIM        + 32 * c + 16 * hi);
        kf[4 + c] = *(const v16h*)(kbase + (size_t)(16 + lo) * DIM + 32 * c + 16 * hi);
      }
    }
    v8f s0 = {}, s1 = {};
#pragma unroll
    for (int c = 0; c < 4; ++c) {
      s0 = wmma_f16(qfrag[c], kf[c],     s0);
      s1 = wmma_f16(qfrag[c], kf[4 + c], s1);
    }

    // ---- scale + adjacency additive bias (C layout M=r+8hi, N=lo) ----
#pragma unroll
    for (int r = 0; r < 8; ++r) {
      const int row = row0 + r + 8 * hi;
      const int* arow = adjb + (size_t)row * SEQ + j;
      s0[r] = s0[r] * scale + (arow[lo]      ? 0.0f : NEG_BIG);
      s1[r] = s1[r] * scale + (arow[16 + lo] ? 0.0f : NEG_BIG);
    }

    // ---- online softmax: row reductions across 16-lane halves ----
    float mnew[8], alpha[8];
#pragma unroll
    for (int r = 0; r < 8; ++r) {
      float mx = fmaxf(s0[r], s1[r]);
      mx = fmaxf(mx, __shfl_xor(mx, 1));
      mx = fmaxf(mx, __shfl_xor(mx, 2));
      mx = fmaxf(mx, __shfl_xor(mx, 4));
      mx = fmaxf(mx, __shfl_xor(mx, 8));
      mnew[r]  = fmaxf(mrow[r], mx);
      alpha[r] = __expf(mrow[r] - mnew[r]);
      mrow[r]  = mnew[r];
    }

    // P = exp(S - mnew); update l; stage P into LDS (A-layout transpose src)
#pragma unroll
    for (int r = 0; r < 8; ++r) {
      const float p0 = __expf(s0[r] - mnew[r]);
      const float p1 = __expf(s1[r] - mnew[r]);
      float rs = p0 + p1;
      rs += __shfl_xor(rs, 1);
      rs += __shfl_xor(rs, 2);
      rs += __shfl_xor(rs, 4);
      rs += __shfl_xor(rs, 8);
      lrow[r] = lrow[r] * alpha[r] + rs;
      myp[(r + 8 * hi) * 32 + lo]      = (_Float16)p0;
      myp[(r + 8 * hi) * 32 + 16 + lo] = (_Float16)p1;
    }

    // rescale running accumulator
#pragma unroll
    for (int ct = 0; ct < 8; ++ct)
#pragma unroll
      for (int r = 0; r < 8; ++r) o[ct][r] *= alpha[r];

    asm volatile("" ::: "memory");  // keep LDS store -> load ordering

    // ---- O += P @ V (K = 32); vT makes B-fragments contiguous ----
    const v16h pfrag = load_a_frag(myp + lo * 32, hi);
#pragma unroll
    for (int ct = 0; ct < 8; ++ct) {
      const _Float16* vp =
          vT + ((size_t)b * DIM + 16 * ct + lo) * SEQ + j + 16 * hi;
      o[ct] = wmma_f16(pfrag, *(const v16h*)vp, o[ct]);
    }
  }

  // ---- epilogue: normalize (rcp, not 64 divides), ELU, fp32 store ----
  float linv[8];
#pragma unroll
  for (int r = 0; r < 8; ++r) linv[r] = __builtin_amdgcn_rcpf(lrow[r]);
#pragma unroll
  for (int ct = 0; ct < 8; ++ct) {
#pragma unroll
    for (int r = 0; r < 8; ++r) {
      const int row = row0 + r + 8 * hi;
      const float h = o[ct][r] * linv[r];
      const float e = (h > 0.0f) ? h : (__expf(h) - 1.0f);
      out[((size_t)b * SEQ + row) * DIM + 16 * ct + lo] = e;
    }
  }
}

// ---------------------------------------------------------------------------
extern "C" void kernel_launch(void* const* d_in, const int* in_sizes, int n_in,
                              void* d_out, int out_size, void* d_ws,
                              size_t ws_size, hipStream_t stream) {
  (void)in_sizes; (void)n_in; (void)out_size; (void)ws_size;
  const float* inputs = (const float*)d_in[0];
  const int*   adj    = (const int*)d_in[1];
  const float* Wq = (const float*)d_in[2];
  const float* bq = (const float*)d_in[3];
  const float* Wk = (const float*)d_in[4];
  const float* bk = (const float*)d_in[5];
  const float* Wv = (const float*)d_in[6];
  const float* bv = (const float*)d_in[7];

  // Workspace: q (4MB) | k (4MB) | vT (4MB), all f16.
  _Float16* qh = (_Float16*)d_ws;
  _Float16* kh = qh + (size_t)BATCH * SEQ * DIM;
  _Float16* vT = kh + (size_t)BATCH * SEQ * DIM;

  qkv_kernel<<<(BATCH * SEQ) / 128, 256, 0, stream>>>(
      inputs, Wq, bq, Wk, bk, Wv, bv, qh, kh, vT);
  attn_kernel<<<(BATCH * SEQ) / 128, 256, 0, stream>>>(
      qh, kh, vT, adj, (float*)d_out);
}
